// MessagePassing_42863773614803
// MI455X (gfx1250) — compile-verified, matrix operand order checked by reading, not run on hardware
//
#include <hip/hip_runtime.h>
#include <hip/hip_bf16.h>

// Problem constants (match reference)
#define B_    8
#define N_    2048
#define DEG_  16
#define DIMH  128
#define MSGD  128
#define INDIM 256   // 2*DIMH

typedef __attribute__((ext_vector_type(16))) __bf16 v16bf;
typedef __attribute__((ext_vector_type(8)))  float  v8f;

union Frag {
    unsigned u[8];
    v16bf    v;
};

__device__ __forceinline__ unsigned short f2bf(float f) {
    // round-to-nearest-even fp32 -> bf16
    unsigned u = __float_as_uint(f);
    unsigned r = u + 0x7FFFu + ((u >> 16) & 1u);
    return (unsigned short)(r >> 16);
}

__device__ __forceinline__ unsigned pack2(float lo, float hi) {
    return (unsigned)f2bf(lo) | ((unsigned)f2bf(hi) << 16);
}

// K offset within a 32-wide k-tile for VGPR v of a 16-bit A/B fragment.
// ISA 7.12.2: VGPR 0-3 hold K = {0..7}+8*half, VGPR 4-7 hold K = {16..23}+8*half.
__device__ __forceinline__ int kfrag(int v, int hf) {
    return ((v < 4) ? (2 * v) : (16 + 2 * (v - 4))) + hf * 8;
}

// ---------------------------------------------------------------------------
// Kernel 0: convert W_msg, W_ih, W_hh, Ht from f32 to bf16 into workspace.
// Workspace element layout (bf16):
//   [0, 32768)                         W_msg  (128 x 256)
//   [32768, 81920)                     W_ih   (384 x 128)
//   [81920, 131072)                    W_hh   (384 x 128)
//   [131072, 2228224)                  Ht     (8 x 2048 x 128)
//   [2228224, 4325376)                 agg    (8 x 2048 x 128)  (written by k1)
// ---------------------------------------------------------------------------
__global__ __launch_bounds__(256) void to_bf16_kernel(
    const float* __restrict__ Ht, const float* __restrict__ Wm,
    const float* __restrict__ Wi, const float* __restrict__ Wh,
    unsigned short* __restrict__ ws)
{
    const int NW  = MSGD * INDIM;       // 32768
    const int NI  = 3 * DIMH * DIMH;    // 49152
    const int NHT = B_ * N_ * DIMH;     // 2097152
    int i = blockIdx.x * 256 + threadIdx.x;
    if (i < NW)                          ws[i] = f2bf(Wm[i]);
    else if (i < NW + NI)                ws[i] = f2bf(Wi[i - NW]);
    else if (i < NW + 2 * NI)            ws[i] = f2bf(Wh[i - NW - NI]);
    else if (i < NW + 2 * NI + NHT)      ws[i] = f2bf(Ht[i - NW - 2 * NI]);
}

// ---------------------------------------------------------------------------
// Kernel 1: per (b,n): gather 16 edges, LayerNorm(256), msg = leaky(xn @ W^T + b),
// mean over the 16 edges (edge_src groups edges by source node) -> agg (bf16).
// WG = 256 threads = 8 waves. LDS holds the 16x256 bf16 A-matrix (stride 264).
// ---------------------------------------------------------------------------
__global__ __launch_bounds__(256) void edge_msg_agg_kernel(
    const float* __restrict__ Ht, const float* __restrict__ gamma,
    const float* __restrict__ beta, const float* __restrict__ b_msg,
    const int* __restrict__ edge_dst,
    const unsigned short* __restrict__ wmsg_bf,   // [128][256] bf16
    unsigned short* __restrict__ agg_bf)          // [B][N][128] bf16
{
    const int n = blockIdx.x;
    const int b = blockIdx.y;
    const int t = threadIdx.x;

    __shared__ unsigned short xbf[16 * 264];      // 16 edges x 256 (stride 264)

    // ---- Stage 1: gather + LayerNorm (16 threads per edge, 16 elems/thread)
    {
        const int e  = t >> 4;          // edge 0..15
        const int g  = t & 15;          // slot in edge group (half-wave)
        const int c0 = g * 16;          // column base (multiple of 16)
        const int dst = edge_dst[n * DEG_ + e];
        const float* srcrow = Ht + ((size_t)b * N_ + n)   * DIMH;
        const float* dstrow = Ht + ((size_t)b * N_ + dst) * DIMH;
        const float* rowp = (c0 < DIMH) ? (srcrow + c0) : (dstrow + (c0 - DIMH));

        float x[16];
        const float4* r4 = (const float4*)rowp;
        #pragma unroll
        for (int q = 0; q < 4; ++q) {
            float4 f = r4[q];
            x[q * 4 + 0] = f.x; x[q * 4 + 1] = f.y;
            x[q * 4 + 2] = f.z; x[q * 4 + 3] = f.w;
        }

        float s = 0.f, sq = 0.f;
        #pragma unroll
        for (int i = 0; i < 16; ++i) { s += x[i]; sq += x[i] * x[i]; }
        // reduce within the 16-lane half-wave group
        #pragma unroll
        for (int m = 1; m < 16; m <<= 1) {
            s  += __shfl_xor(s,  m, 32);
            sq += __shfl_xor(sq, m, 32);
        }
        const float mu  = s  * (1.0f / INDIM);
        const float var = sq * (1.0f / INDIM) - mu * mu;
        const float inv = rsqrtf(var + 1e-5f);

        #pragma unroll
        for (int i = 0; i < 16; i += 2) {
            const int c = c0 + i;
            float y0 = (x[i]     - mu) * inv * gamma[c]     + beta[c];
            float y1 = (x[i + 1] - mu) * inv * gamma[c + 1] + beta[c + 1];
            *(unsigned*)&xbf[e * 264 + c] = pack2(y0, y1);
        }
    }
    __syncthreads();

    // ---- Stage 2: WMMA. Wave w owns msg columns [16w, 16w+16). M=16 edges, K=256.
    const int w    = t >> 5;
    const int lane = t & 31;
    const int hf   = lane >> 4;
    const int ln   = lane & 15;

    v8f acc = {};
    const unsigned short* brow = wmsg_bf + (size_t)(w * 16 + ln) * INDIM; // W_msg row = msg col
    #pragma unroll
    for (int kb = 0; kb < INDIM; kb += 32) {
        Frag A, Bm;
        #pragma unroll
        for (int v = 0; v < 8; ++v) {
            const int k = kb + kfrag(v, hf);
            A.u[v]  = *(const unsigned*)&xbf[ln * 264 + k];   // A[m=ln][k..k+1]
            Bm.u[v] = *(const unsigned*)&brow[k];             // B[k..k+1][n=ln]
        }
        acc = __builtin_amdgcn_wmma_f32_16x16x32_bf16(
                  false, A.v, false, Bm.v, (short)0, acc, false, false);
    }

    // ---- bias + leaky-relu per element, then mean over the 16 edge rows.
    const float bms = b_msg[w * 16 + ln];
    float colsum = 0.f;
    #pragma unroll
    for (int r = 0; r < 8; ++r) {           // element (M = r + 8*hf, N = ln)
        float y = acc[r] + bms;
        y = (y >= 0.f) ? y : 0.2f * y;
        colsum += y;
    }
    colsum += __shfl_xor(colsum, 16, 32);   // combine the two M-halves
    if (hf == 0) {
        agg_bf[((size_t)b * N_ + n) * MSGD + w * 16 + ln] = f2bf(colsum * (1.0f / DEG_));
    }
}

// ---------------------------------------------------------------------------
// Kernel 2: per (b, 16-node tile): gx = agg@W_ih^T, gh = Ht@W_hh^T via WMMA,
// then fused GRU gating -> out. LDS gx/gh: [2][16][388] f32 (padded stride).
// ---------------------------------------------------------------------------
__global__ __launch_bounds__(256) void gru_update_kernel(
    const float* __restrict__ Ht, const float* __restrict__ b_ih,
    const float* __restrict__ b_hh,
    const unsigned short* __restrict__ wih_bf,   // [384][128] bf16
    const unsigned short* __restrict__ whh_bf,   // [384][128] bf16
    const unsigned short* __restrict__ ht_bf,    // [B][N][128] bf16
    const unsigned short* __restrict__ agg_bf,   // [B][N][128] bf16
    float* __restrict__ out)
{
    const int tile  = blockIdx.x;      // 0..127 (node tiles of 16)
    const int b     = blockIdx.y;
    const int t     = threadIdx.x;
    const int node0 = tile * 16;

    __shared__ float lds[2 * 16 * 388];

    const int w    = t >> 5;
    const int lane = t & 31;
    const int hf   = lane >> 4;
    const int ln   = lane & 15;

    // Wave w computes column-tiles {3w, 3w+1, 3w+2} of the 384-wide gx and gh.
    #pragma unroll
    for (int j = 0; j < 3; ++j) {
        const int ct  = w * 3 + j;         // 0..23
        const int col = ct * 16 + ln;      // 0..383
        #pragma unroll
        for (int which = 0; which < 2; ++which) {  // 0: gx(agg,W_ih) 1: gh(Ht,W_hh)
            const unsigned short* abase =
                (which == 0 ? agg_bf : ht_bf) + ((size_t)b * N_ + node0 + ln) * DIMH;
            const unsigned short* brow =
                (which == 0 ? wih_bf : whh_bf) + (size_t)col * DIMH;
            v8f acc = {};
            #pragma unroll
            for (int kb = 0; kb < DIMH; kb += 32) {
                Frag A, Bm;
                #pragma unroll
                for (int v = 0; v < 8; ++v) {
                    const int k = kb + kfrag(v, hf);
                    A.u[v]  = *(const unsigned*)&abase[k];
                    Bm.u[v] = *(const unsigned*)&brow[k];
                }
                acc = __builtin_amdgcn_wmma_f32_16x16x32_bf16(
                          false, A.v, false, Bm.v, (short)0, acc, false, false);
            }
            #pragma unroll
            for (int r = 0; r < 8; ++r) {
                const int m = r + hf * 8;
                lds[which * (16 * 388) + m * 388 + col] = acc[r];
            }
        }
    }
    __syncthreads();

    // ---- fused GRU gating: 16 nodes x 128 dims = 2048 elems, 8 per thread.
    #pragma unroll
    for (int i = 0; i < 8; ++i) {
        const int idx = t + i * 256;       // 0..2047
        const int m = idx >> 7;            // node in tile
        const int d = idx & 127;           // hidden dim
        const float* gx = lds + m * 388;
        const float* gh = lds + 16 * 388 + m * 388;
        const float xr = gx[d]       + b_ih[d];
        const float xz = gx[d + 128] + b_ih[d + 128];
        const float xn = gx[d + 256] + b_ih[d + 256];
        const float hr = gh[d]       + b_hh[d];
        const float hz = gh[d + 128] + b_hh[d + 128];
        const float hn = gh[d + 256] + b_hh[d + 256];
        const float r  = 1.0f / (1.0f + __expf(-(xr + hr)));
        const float z  = 1.0f / (1.0f + __expf(-(xz + hz)));
        const float nn = tanhf(xn + r * hn);
        const size_t o = ((size_t)b * N_ + node0 + m) * DIMH + d;
        out[o] = (1.0f - z) * nn + z * Ht[o];
    }
}

// ---------------------------------------------------------------------------
extern "C" void kernel_launch(void* const* d_in, const int* in_sizes, int n_in,
                              void* d_out, int out_size, void* d_ws, size_t ws_size,
                              hipStream_t stream)
{
    (void)in_sizes; (void)n_in; (void)out_size; (void)ws_size;

    const float* Ht    = (const float*)d_in[0];
    const float* gamma = (const float*)d_in[1];
    const float* beta  = (const float*)d_in[2];
    const float* Wmsg  = (const float*)d_in[3];
    const float* bmsg  = (const float*)d_in[4];
    const float* Wih   = (const float*)d_in[5];
    const float* Whh   = (const float*)d_in[6];
    const float* bih   = (const float*)d_in[7];
    const float* bhh   = (const float*)d_in[8];
    /* edge_src = d_in[9] is implied by layout (edges grouped by src) */
    const int*   edst  = (const int*)d_in[10];
    float* out = (float*)d_out;

    unsigned short* ws = (unsigned short*)d_ws;
    unsigned short* wmsg_bf = ws;                                   // 32768
    unsigned short* wih_bf  = ws + 32768;                           // 49152
    unsigned short* whh_bf  = ws + 32768 + 49152;                   // 49152
    unsigned short* ht_bf   = ws + 32768 + 2 * 49152;               // 2097152
    unsigned short* agg_bf  = ht_bf + (size_t)B_ * N_ * DIMH;       // 2097152

    // 32768 + 98304 + 2097152 = 2,228,224 elements to convert
    to_bf16_kernel<<<(2228224 + 255) / 256, 256, 0, stream>>>(Ht, Wmsg, Wih, Whh, ws);

    edge_msg_agg_kernel<<<dim3(N_, B_), 256, 0, stream>>>(
        Ht, gamma, beta, bmsg, edst, wmsg_bf, agg_bf);

    gru_update_kernel<<<dim3(N_ / 16, B_), 256, 0, stream>>>(
        Ht, bih, bhh, wih_bf, whh_bf, ht_bf, agg_bf, out);
}
